// BitLinear_41403484733545
// MI455X (gfx1250) — 8-bit matrix operand order repaired and checked by reading, compiled, not run on hardware
//
#include <hip/hip_runtime.h>
#include <stdint.h>


typedef __attribute__((ext_vector_type(8))) int v8i;

#define MDIM   8192
#define NDIM   8192
#define KDIM   2048
#define KTILES (KDIM / 64)
#define WELEMS ((size_t)NDIM * KDIM)

#define PART_OFF   0u
#define GAMMA_OFF  4096u
#define ALPHA_OFF  8192u
#define APACK_OFF  (1u << 16)
#define BPACK_OFF  ((1u << 16) + (1u << 24))

__device__ __forceinline__ int ka_start(int v, int g) {
    return ((v >> 2) << 5) + (g << 4) + ((v & 3) << 2);
}
__device__ __forceinline__ int kb_start(int v, int g) {
    return ((v >> 2) << 5) + (g << 4) + ((v & 3) << 2);
}

__global__ __launch_bounds__(256) void reduce_abs_partial(
    const float* __restrict__ w, float* __restrict__ partials) {
    __shared__ float red[256];
    size_t base = (size_t)blockIdx.x * 16384u;
    float s = 0.0f;
    for (int i = 0; i < 64; ++i)
        s += fabsf(w[base + threadIdx.x + (size_t)i * 256u]);
    red[threadIdx.x] = s;
    __syncthreads();
    for (int off = 128; off > 0; off >>= 1) {
        if ((int)threadIdx.x < off) red[threadIdx.x] += red[threadIdx.x + off];
        __syncthreads();
    }
    if (threadIdx.x == 0) partials[blockIdx.x] = red[0];
}

__global__ __launch_bounds__(256) void reduce_final(
    const float* __restrict__ partials, float* __restrict__ gamma_out) {
    __shared__ float red[256];
    float s = partials[threadIdx.x] + partials[threadIdx.x + 256] +
              partials[threadIdx.x + 512] + partials[threadIdx.x + 768];
    red[threadIdx.x] = s;
    __syncthreads();
    for (int off = 128; off > 0; off >>= 1) {
        if ((int)threadIdx.x < off) red[threadIdx.x] += red[threadIdx.x + off];
        __syncthreads();
    }
    if (threadIdx.x == 0)
        gamma_out[0] = red[0] * (1.0f / (float)WELEMS);
}

__global__ __launch_bounds__(256) void quant_w(
    const float* __restrict__ w, const float* __restrict__ gamma_p,
    uint32_t* __restrict__ bpack) {
    uint32_t t   = blockIdx.x * 256u + threadIdx.x;
    uint32_t idx = t & 255u;
    uint32_t tile = t >> 8;
    int kt = (int)(tile & (KTILES - 1));
    int nt = (int)(tile / KTILES);
    int lane = (int)(idx >> 3);
    int v    = (int)(idx & 7);
    int n = nt * 16 + (lane & 15);
    int g = lane >> 4;
    int k = kt * 64 + kb_start(v, g);
    float inv = 1.0f / fmaxf(gamma_p[0], 1e-5f);
    const float* wr = w + (size_t)n * KDIM + k;
    uint32_t dw = 0;
    #pragma unroll
    for (int b = 0; b < 4; ++b) {
        int q = (int)rintf(wr[b] * inv);
        q = q < -1 ? -1 : (q > 1 ? 1 : q);
        dw |= ((uint32_t)(uint8_t)(int8_t)q) << (8 * b);
    }
    bpack[t] = dw;
}

__global__ __launch_bounds__(256) void quant_x(
    const float* __restrict__ x, uint32_t* __restrict__ apack,
    float* __restrict__ alpha) {
    __shared__ float red[256];
    int m = blockIdx.x;
    const float* xr = x + (size_t)m * KDIM;
    float mx = 0.0f;
    for (int i = threadIdx.x; i < KDIM; i += 256)
        mx = fmaxf(mx, fabsf(xr[i]));
    red[threadIdx.x] = mx;
    __syncthreads();
    for (int off = 128; off > 0; off >>= 1) {
        if ((int)threadIdx.x < off)
            red[threadIdx.x] = fmaxf(red[threadIdx.x], red[threadIdx.x + off]);
        __syncthreads();
    }
    float a = red[0];
    if (threadIdx.x == 0) alpha[m] = a;
    float inv = 127.0f / fmaxf(a, 1e-5f);
    int mt = m >> 4;
    int ml = m & 15;
    for (int d = threadIdx.x; d < 512; d += 256) {
        int kt = d >> 4;
        int g  = (d >> 3) & 1;
        int v  = d & 7;
        int k  = kt * 64 + ka_start(v, g);
        uint32_t dw = 0;
        #pragma unroll
        for (int b = 0; b < 4; ++b) {
            int q = (int)rintf(xr[k + b] * inv);
            q = q < -127 ? -127 : (q > 127 ? 127 : q);
            dw |= ((uint32_t)(uint8_t)(int8_t)q) << (8 * b);
        }
        size_t off = ((size_t)mt * KTILES + kt) * 256u + (size_t)(ml + (g << 4)) * 8u + v;
        apack[off] = dw;
    }
}

__global__ __launch_bounds__(256) void gemm_iu8(
    const char* __restrict__ apack, const char* __restrict__ bpack,
    const float* __restrict__ alpha, const float* __restrict__ gamma_p,
    float* __restrict__ out) {
    int lane = threadIdx.x & 31;
    int wave = threadIdx.x >> 5;
    int wm = wave >> 2;
    int wn = wave & 3;
    int mt0 = blockIdx.y * 8 + wm * 4;
    int nt0 = blockIdx.x * 16 + wn * 4;

    v8i acc[4][4];
    #pragma unroll
    for (int i = 0; i < 4; ++i)
        #pragma unroll
        for (int j = 0; j < 4; ++j)
            acc[i][j] = (v8i)0;

    size_t lb = (size_t)lane << 5;
    for (int kt = 0; kt < KTILES; ++kt) {
        v8i a[4], b[4];
        #pragma unroll
        for (int i = 0; i < 4; ++i)
            a[i] = *(const v8i*)(apack + ((((size_t)(mt0 + i) * KTILES) + kt) << 10) + lb);
        #pragma unroll
        for (int j = 0; j < 4; ++j)
            b[j] = *(const v8i*)(bpack + ((((size_t)(nt0 + j) * KTILES) + kt) << 10) + lb);
        #pragma unroll
        for (int i = 0; i < 4; ++i)
            #pragma unroll
            for (int j = 0; j < 4; ++j)
                acc[i][j] = __builtin_amdgcn_wmma_i32_16x16x64_iu8(
                     true, a[i],  true, b[j],
                    acc[i][j],  false,  false);
    }

    float gscale = gamma_p[0] * (1.0f / 127.0f);
    int nl   = lane & 15;
    int half = lane >> 4;
    #pragma unroll
    for (int i = 0; i < 4; ++i) {
        #pragma unroll
        for (int r = 0; r < 8; ++r) {
            int m = (mt0 + i) * 16 + half * 8 + r;
            float s = alpha[m] * gscale;
            #pragma unroll
            for (int j = 0; j < 4; ++j) {
                int n = (nt0 + j) * 16 + nl;
                __builtin_nontemporal_store((float)acc[i][j][r] * s,
                                            &out[(size_t)m * NDIM + n]);
            }
        }
    }
}

extern "C" void kernel_launch(void* const* d_in, const int* in_sizes, int n_in,
                              void* d_out, int out_size, void* d_ws, size_t ws_size,
                              hipStream_t stream) {
    const float* x = (const float*)d_in[0];
    const float* w = (const float*)d_in[1];
    float* out = (float*)d_out;

    char* ws = (char*)d_ws;
    float*    partials = (float*)(ws + PART_OFF);
    float*    gamma    = (float*)(ws + GAMMA_OFF);
    float*    alpha    = (float*)(ws + ALPHA_OFF);
    uint32_t* apack    = (uint32_t*)(ws + APACK_OFF);
    uint32_t* bpack    = (uint32_t*)(ws + BPACK_OFF);

    reduce_abs_partial<<<1024, 256, 0, stream>>>(w, partials);
    reduce_final<<<1, 256, 0, stream>>>(partials, gamma);

    quant_w<<<(WELEMS / 4) / 256, 256, 0, stream>>>(w, gamma, bpack);

    quant_x<<<MDIM, 256, 0, stream>>>(x, apack, alpha);

    dim3 grid(NDIM / 256, MDIM / 128);
    gemm_iu8<<<grid, 256, 0, stream>>>((const char*)apack, (const char*)bpack,
                                       alpha, gamma, out);
}
